// FasterRCNN_30468497998476
// MI455X (gfx1250) — matrix operand order repaired and checked by reading, not compiled, not run on hardware
//
#include <hip/hip_runtime.h>

// ---------------- problem constants ----------------
#define B2       2
#define CIN      256
#define FH       50
#define FW       50
#define CMID     512
#define K1       2304          // 256*3*3
#define NSP      5000          // 2 * 50*50
#define NPAD     5056          // padded to multiple of 64
#define M2P      80            // 36 loc (pad to 48) + 18 obj (48..65) pad to 80
#define NA       22500         // anchors per batch
#define NPRE     6000
#define NPOST    300
#define SORTN    32768

typedef __attribute__((ext_vector_type(16))) __bf16 v16bf;
typedef __attribute__((ext_vector_type(8)))  __bf16 v8bf;
typedef __attribute__((ext_vector_type(8)))  float  v8f;

// ---------------- pack conv1 weights f32 -> bf16 [512][2304] ----------------
__global__ void pack_w1(const float* __restrict__ w, __bf16* __restrict__ Wb) {
  int i = blockIdx.x * blockDim.x + threadIdx.x;
  if (i < CMID * K1) Wb[i] = (__bf16)w[i];
}

// ---- pack loc/obj 1x1 weights into Wb2[80][512] bf16 (+bias2[80] f32) ----
__global__ void pack_w2(const float* __restrict__ locw, const float* __restrict__ objw,
                        const float* __restrict__ locb, const float* __restrict__ objb,
                        __bf16* __restrict__ Wb2, float* __restrict__ bias2) {
  int i = blockIdx.x * blockDim.x + threadIdx.x;
  if (i >= M2P * CMID) return;
  int m = i / CMID, k = i % CMID;
  float v = 0.f;
  if (m < 36) v = locw[m * CMID + k];
  else if (m >= 48 && m < 66) v = objw[(m - 48) * CMID + k];
  Wb2[i] = (__bf16)v;
  if (i < M2P) {
    float bv = 0.f;
    if (i < 36) bv = locb[i];
    else if (i >= 48 && i < 66) bv = objb[i - 48];
    bias2[i] = bv;
  }
}

// ------- im2col + f32->bf16: Xc[NPAD][K1], n = b*2500 + oh*50 + ow -------
__global__ void im2col_k(const float* __restrict__ x, __bf16* __restrict__ Xc) {
  long long i = (long long)blockIdx.x * blockDim.x + threadIdx.x;
  if (i >= (long long)NPAD * K1) return;
  int n = (int)(i / K1), k = (int)(i % K1);
  float v = 0.f;
  if (n < NSP) {
    int b = n / (FH * FW), cell = n % (FH * FW);
    int oh = cell / FW, ow = cell % FW;
    int c = k / 9, r = k % 9, kh = r / 3, kw = r % 3;
    int ih = oh + kh - 1, iw = ow + kw - 1;
    if (ih >= 0 && ih < FH && iw >= 0 && iw < FW)
      v = x[(((long long)b * CIN + c) * FH + ih) * FW + iw];
  }
  Xc[i] = (__bf16)v;
}

// helper: assemble 16-bit A fragment (16x32) from two contiguous 8-elem runs
__device__ __forceinline__ v16bf load_a_frag(const __bf16* row, int k, int half) {
  v8bf alo = *(const v8bf*)(row + k + half * 16);
  v8bf ahi = *(const v8bf*)(row + k + half * 16 + 8);
  v16bf a;
#pragma unroll
  for (int e = 0; e < 8; ++e) { a[e] = alo[e]; a[8 + e] = ahi[e]; }
  return a;
}

// ---- GEMM1, 64x64 macro-tile per wave: 16 WMMAs / 8 fragment loads per K-step ----
__global__ __launch_bounds__(32)
void gemm1_wmma(const __bf16* __restrict__ Wb, const __bf16* __restrict__ Xc,
                const float* __restrict__ bias, __bf16* __restrict__ Hb) {
  const int l = threadIdx.x;
  const int half = l >> 4, lm = l & 15;
  const __bf16* arow[4];
  const __bf16* brow[4];
#pragma unroll
  for (int i = 0; i < 4; ++i) {
    arow[i] = Wb + (long long)(blockIdx.x * 64 + i * 16 + lm) * K1;
    brow[i] = Xc + (long long)(blockIdx.y * 64 + i * 16 + lm) * K1;
  }
  v8f acc[4][4] = {};
  for (int k = 0; k < K1; k += 32) {
    if (k + 128 < K1) __builtin_prefetch(arow[0] + k + 128, 0, 0);
    v16bf a[4], b[4];
#pragma unroll
    for (int i = 0; i < 4; ++i) {
      a[i] = load_a_frag(arow[i], k, half);
      b[i] = *(const v16bf*)(brow[i] + k + half * 16);
    }
#pragma unroll
    for (int mi = 0; mi < 4; ++mi)
#pragma unroll
      for (int ni = 0; ni < 4; ++ni)
        acc[mi][ni] = __builtin_amdgcn_wmma_f32_16x16x32_bf16(
            false, a[mi], false, b[ni], (short)0, acc[mi][ni], false, false);
  }
  // C/D layout: lane -> col = l%16, rows = 8*(l/16)+r
#pragma unroll
  for (int mi = 0; mi < 4; ++mi) {
    const int mb = blockIdx.x * 64 + mi * 16 + half * 8;
#pragma unroll
    for (int ni = 0; ni < 4; ++ni) {
      const int outn = blockIdx.y * 64 + ni * 16 + lm;
      __bf16* dst = Hb + (long long)outn * CMID + mb;
#pragma unroll
      for (int r = 0; r < 8; ++r) {
        float v = acc[mi][ni][r] + bias[mb + r];
        dst[r] = (__bf16)(v > 0.f ? v : 0.f);
      }
    }
  }
}

// ---- GEMM2, 16x64 macro-tile: A fragment reused across 4 N-tiles, f32 out ----
__global__ __launch_bounds__(32)
void gemm2_wmma(const __bf16* __restrict__ Wb2, const __bf16* __restrict__ Hb,
                const float* __restrict__ bias2, float* __restrict__ O) {
  const int l = threadIdx.x;
  const int half = l >> 4, lm = l & 15;
  const __bf16* arow = Wb2 + (long long)(blockIdx.x * 16 + lm) * CMID;
  const __bf16* brow[4];
#pragma unroll
  for (int i = 0; i < 4; ++i)
    brow[i] = Hb + (long long)(blockIdx.y * 64 + i * 16 + lm) * CMID;
  v8f acc[4] = {};
  for (int k = 0; k < CMID; k += 32) {
    v16bf a = load_a_frag(arow, k, half);
#pragma unroll
    for (int ni = 0; ni < 4; ++ni) {
      v16bf b = *(const v16bf*)(brow[ni] + k + half * 16);
      acc[ni] = __builtin_amdgcn_wmma_f32_16x16x32_bf16(
          false, a, false, b, (short)0, acc[ni], false, false);
    }
  }
  const int mb = blockIdx.x * 16 + half * 8;
#pragma unroll
  for (int ni = 0; ni < 4; ++ni) {
    const int outn = blockIdx.y * 64 + ni * 16 + lm;
    float* dst = O + (long long)outn * M2P + mb;
#pragma unroll
    for (int r = 0; r < 8; ++r) dst[r] = acc[ni][r] + bias2[mb + r];
  }
}

// ------------- decode boxes + scores + validity -------------
__global__ void decode_k(const float* __restrict__ O, const float* __restrict__ anch,
                         const int* __restrict__ imh, const int* __restrict__ imw,
                         float* __restrict__ boxes, float* __restrict__ svals) {
  int i = blockIdx.x * blockDim.x + threadIdx.x;
  if (i >= B2 * NA) return;
  int b = i / NA, p = i % NA;
  int cell = p / 9, a = p % 9;
  const float* row = O + (long long)(b * (FH * FW) + cell) * M2P;
  float lx = row[a * 4 + 0], ly = row[a * 4 + 1];
  float lw = row[a * 4 + 2], lh = row[a * 4 + 3];
  float o0 = row[48 + a * 2], o1 = row[48 + a * 2 + 1];
  float score = 1.f / (1.f + __expf(o0 - o1));
  const float* an = anch + (long long)p * 4;
  float aw = an[2] - an[0], ah = an[3] - an[1];
  float acx = an[0] + 0.5f * aw, acy = an[1] + 0.5f * ah;
  float cx = acx + lx * aw, cy = acy + ly * ah;
  float w = aw * __expf(lw), h = ah * __expf(lh);
  float W = (float)imw[0], H = (float)imh[0];
  float x1 = fminf(fmaxf(cx - 0.5f * w, 0.f), W);
  float y1 = fminf(fmaxf(cy - 0.5f * h, 0.f), H);
  float x2 = fminf(fmaxf(cx + 0.5f * w, 0.f), W);
  float y2 = fminf(fmaxf(cy + 0.5f * h, 0.f), H);
  float* bo = boxes + (long long)i * 4;
  bo[0] = x1; bo[1] = y1; bo[2] = x2; bo[3] = y2;
  bool valid = (x2 - x1 >= 16.f) && (y2 - y1 >= 16.f);
  svals[i] = valid ? score : -1.0f;
}

// ------- per-batch bitonic sort in 256KB LDS, emit top-6000 (desc, stable) -------
__global__ void sort_topk(const float* __restrict__ svals,
                          int* __restrict__ selIdx, float* __restrict__ selScore) {
  extern __shared__ unsigned long long key[];   // SORTN entries = 256 KB
  const int b = blockIdx.x, t = threadIdx.x, nt = blockDim.x;
  for (int i = t; i < SORTN; i += nt) {
    unsigned long long kk = 0ull;
    if (i < NA) {
      unsigned u = __builtin_bit_cast(unsigned, svals[b * NA + i]);
      u = (u & 0x80000000u) ? ~u : (u | 0x80000000u);    // order-preserving map
      kk = ((unsigned long long)u << 32) | (unsigned)(0xFFFFFFFFu - i);
    }
    key[i] = kk;
  }
  __syncthreads();
  for (int ks = 2; ks <= SORTN; ks <<= 1) {
    for (int j = ks >> 1; j >= 1; j >>= 1) {
      for (int i = t; i < SORTN; i += nt) {
        int p = i ^ j;
        if (p > i) {
          bool up = ((i & ks) == 0);                     // ascending overall
          unsigned long long a = key[i], c = key[p];
          if (up ? (a > c) : (a < c)) { key[i] = c; key[p] = a; }
        }
      }
      __syncthreads();
    }
  }
  for (int r = t; r < NPRE; r += nt) {
    unsigned long long kk = key[SORTN - 1 - r];          // descending read-out
    int idx = (int)(0xFFFFFFFFu - (unsigned)(kk & 0xFFFFFFFFull));
    selIdx[b * NPRE + r] = idx;
    selScore[b * NPRE + r] = svals[b * NA + idx];
  }
}

// ------- greedy NMS with boxes resident in LDS, then emit final 300 -------
__global__ void nms_emit(const float* __restrict__ boxes, const int* __restrict__ selIdx,
                         const float* __restrict__ selScore, float* __restrict__ out) {
  extern __shared__ char sm[];
  float4* sb  = (float4*)sm;                       //  96000 B
  int*    kp  = (int*)(sm + 96000);                //  24000 B
  float*  ss  = (float*)(sm + 120000);             //  24000 B
  const int b = blockIdx.x, t = threadIdx.x, nt = blockDim.x;
  for (int i = t; i < NPRE; i += nt) {
    int gi = selIdx[b * NPRE + i];
    const float* bp = boxes + ((long long)b * NA + gi) * 4;
    sb[i] = make_float4(bp[0], bp[1], bp[2], bp[3]);
    ss[i] = selScore[b * NPRE + i];
    kp[i] = 1;
  }
  __syncthreads();
  for (int i = 0; i < NPRE; ++i) {
    if (kp[i]) {
      float4 bi = sb[i];
      float ai = (bi.z - bi.x) * (bi.w - bi.y);
      for (int j = i + 1 + t; j < NPRE; j += nt) {
        if (!kp[j]) continue;
        float4 bj = sb[j];
        float ix1 = fmaxf(bi.x, bj.x), iy1 = fmaxf(bi.y, bj.y);
        float ix2 = fminf(bi.z, bj.z), iy2 = fminf(bi.w, bj.w);
        float inter = fmaxf(ix2 - ix1, 0.f) * fmaxf(iy2 - iy1, 0.f);
        float aj = (bj.z - bj.x) * (bj.w - bj.y);
        float iou = inter / (ai + aj - inter);
        if (iou > 0.7f) kp[j] = 0;
      }
    }
    __syncthreads();
  }
  if (t == 0) {
    float* ob = out + (long long)b * NPOST * 4;          // boxes region (2,300,4)
    float* os = out + 2 * NPOST * 4 + (long long)b * NPOST;     // scores
    float* om = out + 2 * NPOST * 4 + 2 * NPOST + (long long)b * NPOST; // mask
    int cnt = 0;
    for (int i = 0; i < NPRE && cnt < NPOST; ++i) {
      if (kp[i] && ss[i] >= 0.f) {
        float4 bx = sb[i];
        ob[cnt * 4 + 0] = bx.x; ob[cnt * 4 + 1] = bx.y;
        ob[cnt * 4 + 2] = bx.z; ob[cnt * 4 + 3] = bx.w;
        os[cnt] = ss[i]; om[cnt] = 1.0f;
        ++cnt;
      }
    }
    for (int r = cnt; r < NPOST; ++r) {
      ob[r * 4 + 0] = ob[r * 4 + 1] = ob[r * 4 + 2] = ob[r * 4 + 3] = 0.f;
      os[r] = 0.f; om[r] = 0.f;
    }
  }
}

extern "C" void kernel_launch(void* const* d_in, const int* in_sizes, int n_in,
                              void* d_out, int out_size, void* d_ws, size_t ws_size,
                              hipStream_t stream) {
  const float* x      = (const float*)d_in[0];
  const float* c1w    = (const float*)d_in[1];
  const float* c1b    = (const float*)d_in[2];
  const float* objw   = (const float*)d_in[3];
  const float* objb   = (const float*)d_in[4];
  const float* locw   = (const float*)d_in[5];
  const float* locb   = (const float*)d_in[6];
  const float* anch   = (const float*)d_in[7];
  const int*   imh    = (const int*)d_in[8];
  const int*   imw    = (const int*)d_in[9];
  float* out = (float*)d_out;

  char* ws = (char*)d_ws;
  size_t off = 0;
  auto carve = [&](size_t bytes) { char* p = ws + off; off += (bytes + 255) & ~(size_t)255; return p; };
  __bf16* Wb    = (__bf16*)carve((size_t)CMID * K1 * 2);
  __bf16* Wb2   = (__bf16*)carve((size_t)M2P * CMID * 2);
  float*  bias2 = (float*) carve((size_t)M2P * 4);
  __bf16* Xc    = (__bf16*)carve((size_t)NPAD * K1 * 2);
  __bf16* Hb    = (__bf16*)carve((size_t)NPAD * CMID * 2);
  float*  O     = (float*) carve((size_t)NPAD * M2P * 4);
  float*  boxes = (float*) carve((size_t)B2 * NA * 4 * 4);
  float*  svals = (float*) carve((size_t)B2 * NA * 4);
  int*    selI  = (int*)   carve((size_t)B2 * NPRE * 4);
  float*  selS  = (float*) carve((size_t)B2 * NPRE * 4);

  pack_w1<<<(CMID * K1 + 255) / 256, 256, 0, stream>>>(c1w, Wb);
  pack_w2<<<(M2P * CMID + 255) / 256, 256, 0, stream>>>(locw, objw, locb, objb, Wb2, bias2);
  {
    long long n = (long long)NPAD * K1;
    im2col_k<<<(unsigned)((n + 255) / 256), 256, 0, stream>>>(x, Xc);
  }
  gemm1_wmma<<<dim3(CMID / 64, NPAD / 64), 32, 0, stream>>>(Wb, Xc, c1b, Hb);
  gemm2_wmma<<<dim3(M2P / 16, NPAD / 64), 32, 0, stream>>>(Wb2, Hb, bias2, O);
  decode_k<<<(B2 * NA + 255) / 256, 256, 0, stream>>>(O, anch, imh, imw, boxes, svals);
  sort_topk<<<B2, 1024, (size_t)SORTN * 8, stream>>>(svals, selI, selS);
  nms_emit<<<B2, 1024, 144000, stream>>>(boxes, selI, selS, out);
}
